// GAA_5634997093105
// MI455X (gfx1250) — matrix operand order repaired and checked by reading, not compiled, not run on hardware
//
#include <hip/hip_runtime.h>

typedef __attribute__((ext_vector_type(16))) _Float16 v16h;
typedef __attribute__((ext_vector_type(8)))  _Float16 v8h;
typedef __attribute__((ext_vector_type(8)))  float    v8f;

#define WPB  4      // windows per block
#define NTHR 512    // 16 waves (wave32)

__device__ __forceinline__ v16h wmma_pack(v8h lo, v8h hi) {
  v16h r;
#pragma unroll
  for (int i = 0; i < 8; ++i) { r[i] = lo[i]; r[i + 8] = hi[i]; }
  return r;
}

// A fragment (16x32 f16) from LDS row-major [rows][ldk] halves.
// Lane layout (contiguous K): lanes 0-15: M=lane, K = k0+{0..15};
//                      lanes 16-31: M=lane-16, K = k0+{16..31}.
__device__ __forceinline__ v16h frag_a_lds(const _Float16* p, int ldk, int row0,
                                           int k0, int lane) {
  int m  = row0 + (lane & 15);
  int kb = k0 + ((lane >> 4) << 4);
  const _Float16* q = p + m * ldk + kb;
  v8h lo = *(const v8h*)q;
  v8h hi = *(const v8h*)(q + 8);
  return wmma_pack(lo, hi);
}

// B fragment (32x16 f16) from row-major [N][ldk]: column N = lane&15 holds
// contiguous K (lanes 0-15: K=k0..k0+15, lanes 16-31: K=k0+16..k0+31).
__device__ __forceinline__ v16h frag_b_rowmajor(const _Float16* p, int ldk,
                                                int n0, int k0, int lane) {
  int n = n0 + (lane & 15);
  int k = k0 + ((lane >> 4) << 4);
  return *(const v16h*)(p + n * ldk + k);
}

__device__ __forceinline__ v8f wmma_f16(v16h a, v16h b, v8f c) {
  return __builtin_amdgcn_wmma_f32_16x16x32_f16(false, a, false, b, (short)0, c,
                                                false, false);
}

// ---- prep: convert 4x [in=256][out=256] fp32 weights -> f16 transposed
// Wt[out][in] in workspace (contiguous K per output row for B fragments) ----
__global__ void __launch_bounds__(256)
convert_weights_f16t(const float* __restrict__ wq, const float* __restrict__ wk,
                     const float* __restrict__ wv, const float* __restrict__ wo,
                     _Float16* __restrict__ wts) {
  int idx = blockIdx.x * blockDim.x + threadIdx.x;  // 0..262143
  int m = idx >> 16;
  int e = idx & 65535;
  int o = e >> 8;
  int i = e & 255;
  const float* w = (m == 0) ? wq : (m == 1) ? wk : (m == 2) ? wv : wo;
  wts[(size_t)m * 65536 + o * 256 + i] = (_Float16)w[i * 256 + o];
}

// ---- fused windowed attention: QKV proj -> scores -> softmax -> PV -> out proj
// One block handles WPB=4 windows (64 tokens x 256 ch).
// LDS: 136KB (one 32KB region time-shared by X / scores / ctx) -> two
// 16-wave workgroups per 320KB WGP. Phase 1 reuses each X A-fragment
// across Q/K/V; kk loop kept rolled so the scheduler cannot over-prefetch
// weight fragments into spills. ----
__global__ void __launch_bounds__(NTHR)
fused_window_attn(const float* __restrict__ x,
                  const _Float16* __restrict__ wts,
                  const float* __restrict__ bq, const float* __restrict__ bk,
                  const float* __restrict__ bv, const float* __restrict__ bo,
                  float* __restrict__ out) {
  extern __shared__ _Float16 sm[];
  // region 0 (32KB): sX (ph0-1) -> sS scores f32 (ph2-3) -> sC ctx (ph4-5)
  _Float16* sX  = sm;
  float*    sS  = (float*)sm;
  _Float16* sC  = sm;
  _Float16* sQ  = sm + 1 * WPB * 4096; // [WPB*16][256]       32KB
  _Float16* sK  = sm + 2 * WPB * 4096; // [WPB*16][256]       32KB
  _Float16* sVt = sm + 3 * WPB * 4096; // [WPB][256][16] V^T  32KB
  _Float16* sP  = sm + 4 * WPB * 4096; // [WPB*2*16][32] probs 8KB

  const int tid   = threadIdx.x;
  const int lane  = tid & 31;
  const int wave  = tid >> 5;     // 0..15
  const int hw    = lane >> 4;    // half-wave 0/1
  const int lan16 = lane & 15;

  const size_t base = (size_t)blockIdx.x * (WPB * 16 * 256);

  // ---------- Phase 0: X fp32 -> f16 into LDS ----------
  {
    const float* xg = x + base;
    int off = tid * 32;  // 16384 floats / 512 threads
#pragma unroll
    for (int i = 0; i < 4; ++i) {
      float4 a = *(const float4*)(xg + off + i * 8);
      float4 b = *(const float4*)(xg + off + i * 8 + 4);
      v8h h;
      h[0] = (_Float16)a.x; h[1] = (_Float16)a.y;
      h[2] = (_Float16)a.z; h[3] = (_Float16)a.w;
      h[4] = (_Float16)b.x; h[5] = (_Float16)b.y;
      h[6] = (_Float16)b.z; h[7] = (_Float16)b.w;
      *(v8h*)(sX + off + i * 8) = h;
    }
  }
  __syncthreads();

  // ---------- Phase 1: Q/K/V projections ----------
  // Each wave owns one output N-tile (nt = wave) for ALL THREE matrices.
  // Windows processed in pairs: acc[3][2] (48 VGPRs) + 2 A-frags + 3 B-frags
  // per iteration; rolled kk loop bounds in-flight registers (no spills).
  {
    const int n0 = wave * 16;
    float bl[3];
    bl[0] = bq[n0 + lan16];
    bl[1] = bk[n0 + lan16];
    bl[2] = bv[n0 + lan16];
#pragma unroll 1
    for (int wp = 0; wp < 2; ++wp) {
      const int w0 = wp * 2;
      v8f acc[3][2];
#pragma unroll
      for (int mat = 0; mat < 3; ++mat)
#pragma unroll
        for (int wi = 0; wi < 2; ++wi)
#pragma unroll
          for (int g = 0; g < 8; ++g) acc[mat][wi][g] = bl[mat];

#pragma unroll 1
      for (int kk = 0; kk < 8; ++kk) {
        v16h a0 = frag_a_lds(sX, 256, (w0 + 0) * 16, kk * 32, lane);
        v16h a1 = frag_a_lds(sX, 256, (w0 + 1) * 16, kk * 32, lane);
#pragma unroll
        for (int mat = 0; mat < 3; ++mat) {
          v16h b = frag_b_rowmajor(wts + (size_t)mat * 65536, 256, n0,
                                   kk * 32, lane);
          acc[mat][0] = wmma_f16(a0, b, acc[mat][0]);
          acc[mat][1] = wmma_f16(a1, b, acc[mat][1]);
        }
      }

      // store Q/K row-major, V transposed
#pragma unroll
      for (int wi = 0; wi < 2; ++wi) {
        int w = w0 + wi;
#pragma unroll
        for (int g = 0; g < 8; ++g) {
          int m = w * 16 + g + hw * 8;
          sQ[m * 256 + n0 + lan16] = (_Float16)acc[0][wi][g];
          sK[m * 256 + n0 + lan16] = (_Float16)acc[1][wi][g];
        }
        v8h h;
#pragma unroll
        for (int g = 0; g < 8; ++g) h[g] = (_Float16)acc[2][wi][g];
        *(v8h*)(sVt + w * 4096 + (n0 + lan16) * 16 + hw * 8) = h;
      }
    }
  }
  __syncthreads();   // also retires sX before sS overwrites the region

  // ---------- Phase 2: scores = Q . K^T  (8 (window,head) jobs) ----------
  if (wave < WPB * 2) {
    int wi = wave >> 1, h = wave & 1;
    v8f s;
#pragma unroll
    for (int g = 0; g < 8; ++g) s[g] = 0.f;
#pragma unroll
    for (int kk = 0; kk < 4; ++kk) {
      int k0 = h * 128 + kk * 32;
      v16h a = frag_a_lds(sQ, 256, wi * 16, k0, lane);
      v16h b = frag_b_rowmajor(sK, 256, wi * 16, k0, lane);  // B[d][s]=K[s][d]
      s = wmma_f16(a, b, s);
    }
    float* dst = sS + (wi * 2 + h) * 256;
#pragma unroll
    for (int g = 0; g < 8; ++g) dst[(g + hw * 8) * 16 + lan16] = s[g];
  }
  __syncthreads();

  // ---------- Phase 3: softmax (128 rows, f32) ----------
  if (tid < WPB * 2 * 16) {
    const float scale = 0.08838834764831845f;  // 1/sqrt(128)
    const float* src = sS + tid * 16;
    float v[16];
    float mx = -1e30f;
#pragma unroll
    for (int s = 0; s < 16; ++s) { v[s] = src[s] * scale; mx = fmaxf(mx, v[s]); }
    float sum = 0.f;
#pragma unroll
    for (int s = 0; s < 16; ++s) { v[s] = __expf(v[s] - mx); sum += v[s]; }
    float inv = 1.0f / sum;
    _Float16* dst = sP + tid * 32;
#pragma unroll
    for (int s = 0; s < 16; ++s) {
      dst[s] = (_Float16)(v[s] * inv);
      dst[s + 16] = (_Float16)0.f;   // pad K 16..31 with zeros
    }
  }
  __syncthreads();   // retires sS before sC overwrites the region

  // ---------- Phase 4: ctx = P . V  (64 jobs, K=16 padded to 32) ----------
#pragma unroll
  for (int j = 0; j < 4; ++j) {
    int job = wave + j * 16;        // 0..63
    int wi = job >> 4;
    int h  = (job >> 3) & 1;
    int nt = job & 7;
    int n0 = h * 128 + nt * 16;
    v16h a = frag_a_lds(sP, 32, (wi * 2 + h) * 16, 0, lane);
    v16h b = {};                    // upper half-wave: K=16..31 -> zeros
    if (lane < 16) b = *(const v16h*)(sVt + wi * 4096 + (n0 + lan16) * 16);
    v8f c;
#pragma unroll
    for (int g = 0; g < 8; ++g) c[g] = 0.f;
    c = wmma_f16(a, b, c);
#pragma unroll
    for (int g = 0; g < 8; ++g) {
      int m = wi * 16 + g + hw * 8;
      sC[m * 256 + n0 + lan16] = (_Float16)c[g];
    }
  }
  __syncthreads();

  // ---------- Phase 5: out = ctx @ Wo + bo (16 ntile jobs, 1/wave) ----------
  {
    int n0 = wave * 16;
    const _Float16* wt = wts + (size_t)3 * 65536;
    float bl = bo[n0 + lan16];
    v8f acc[WPB];
#pragma unroll
    for (int wi = 0; wi < WPB; ++wi)
#pragma unroll
      for (int g = 0; g < 8; ++g) acc[wi][g] = bl;
#pragma unroll
    for (int kk = 0; kk < 8; ++kk) {
      v16h b = frag_b_rowmajor(wt, 256, n0, kk * 32, lane);
#pragma unroll
      for (int wi = 0; wi < WPB; ++wi) {
        v16h a = frag_a_lds(sC, 256, wi * 16, kk * 32, lane);
        acc[wi] = wmma_f16(a, b, acc[wi]);
      }
    }
    float* og = out + base;
#pragma unroll
    for (int wi = 0; wi < WPB; ++wi)
#pragma unroll
      for (int g = 0; g < 8; ++g) {
        int m = wi * 16 + g + hw * 8;
        og[(size_t)m * 256 + n0 + lan16] = acc[wi][g];
      }
  }
}

extern "C" void kernel_launch(void* const* d_in, const int* in_sizes, int n_in,
                              void* d_out, int out_size, void* d_ws, size_t ws_size,
                              hipStream_t stream) {
  const float* x  = (const float*)d_in[0];
  const float* Wq = (const float*)d_in[1];
  const float* bq = (const float*)d_in[2];
  const float* Wk = (const float*)d_in[3];
  const float* bk = (const float*)d_in[4];
  const float* Wv = (const float*)d_in[5];
  const float* bv = (const float*)d_in[6];
  const float* Wo = (const float*)d_in[7];
  const float* bo = (const float*)d_in[8];
  _Float16* wts = (_Float16*)d_ws;   // 4 * 64K f16 = 512KB

  convert_weights_f16t<<<1024, 256, 0, stream>>>(Wq, Wk, Wv, Wo, wts);

  int windows = in_sizes[0] / (16 * 256);   // 65536
  int blocks  = windows / WPB;              // 16384
  // 4 x 32KB (region0, Q, K, Vt) + 8KB probs = 136KB -> 2 blocks per WGP
  size_t lds_bytes = (size_t)(WPB * 4096) * 4 * sizeof(_Float16)
                   + (size_t)(WPB * 2 * 16 * 32) * sizeof(_Float16);
  fused_window_attn<<<blocks, NTHR, lds_bytes, stream>>>(x, wts, bq, bk, bv, bo,
                                                         (float*)d_out);
}
